// YOLOLayer_63256278335783
// MI455X (gfx1250) — matrix operand order repaired and checked by reading, compiled, not run on hardware
//
#include <hip/hip_runtime.h>
#include <math.h>

// ---------------------------------------------------------------------------
// Problem constants (match reference)
// ---------------------------------------------------------------------------
#define BATCH      16
#define KLAB       50
#define IN_CH      256
#define N_CLASSES  80
#define N_ANCH     3
#define FSIZE      76
#define HW         (FSIZE * FSIZE)          // 5776
#define NTILES     (HW / 16)                // 361
#define N_CH       (5 + N_CLASSES)          // 85
#define OC         (N_ANCH * N_CH)          // 255
#define CELLS      (BATCH * N_ANCH * HW)    // 277248
#define IGNORE_THRE 0.7f

// all 9 anchors in grid units (ANCHORS / STRIDE)
__device__ __constant__ float AGW[9] = {1.25f, 2.0f, 4.125f, 3.75f, 7.75f, 7.375f, 14.5f, 19.5f, 46.625f};
__device__ __constant__ float AGH[9] = {1.625f, 3.75f, 2.875f, 7.625f, 5.625f, 14.875f, 11.25f, 24.75f, 40.75f};
// masked anchors (layer 2 -> indices 0,1,2)
__device__ __constant__ float MW[3] = {1.25f, 2.0f, 4.125f};
__device__ __constant__ float MH[3] = {1.625f, 3.75f, 2.875f};

typedef __attribute__((ext_vector_type(16))) __bf16 v16bf;
typedef __attribute__((ext_vector_type(8)))  __bf16 v8bf;
typedef __attribute__((ext_vector_type(8)))  float  v8f;

// fp32 -> bf16 bits, round-to-nearest-even (used only in one-shot converters)
static __device__ __forceinline__ unsigned short f2bf_bits(float x) {
    union { float f; unsigned u; } v; v.f = x;
    unsigned r = v.u + 0x7FFFu + ((v.u >> 16) & 1u);
    return (unsigned short)(r >> 16);
}

static __device__ __forceinline__ float sigmoidf(float x) {
    return 1.0f / (1.0f + expf(-x));
}

static __device__ __forceinline__ float bce1(float p, float t) {
    float pc = fminf(fmaxf(p, 1e-12f), 1.0f - 1e-7f);
    return -(t * logf(pc) + (1.0f - t) * logf(1.0f - pc));
}

static __device__ __forceinline__ float wave_red(float v) {
    #pragma unroll
    for (int o = 16; o > 0; o >>= 1) v += __shfl_down(v, o, 32);
    return v;
}

// ---------------------------------------------------------------------------
// Kernel 0a: weights fp32 (255x256) -> bf16 (256x256), row 255 zero-padded.
// One-time: 128 KB out; removes all masking/conversion from the GEMM loop.
// ---------------------------------------------------------------------------
__global__ void conv_w_to_bf16(const float* __restrict__ cw, unsigned short* __restrict__ wbf) {
    int tid = blockIdx.x * blockDim.x + threadIdx.x;   // 0 .. 256*256-1
    if (tid >= 256 * IN_CH) return;
    int m = tid / IN_CH;
    float v = (m < OC) ? cw[tid] : 0.0f;
    wbf[tid] = f2bf_bits(v);
}

// ---------------------------------------------------------------------------
// Kernel 0b: activations fp32 (B,C,HW) -> bf16 transposed (B,HW,C).
// Reads coalesced along HW; per-thread contiguous packed-u32 stores.
// One-time 94 MB read + 47 MB write; GEMM B-fragments become one
// contiguous 32-byte load each (K-contiguous).
// ---------------------------------------------------------------------------
__global__ void xin_to_bf16_t(const float* __restrict__ xin, unsigned* __restrict__ xbf32) {
    int tid = blockIdx.x * blockDim.x + threadIdx.x;   // 0 .. B*HW-1
    if (tid >= BATCH * HW) return;
    int b  = tid / HW;
    int hw = tid - b * HW;
    const float* src = xin + (size_t)b * IN_CH * HW + hw;   // stride HW over k
    unsigned* dst = xbf32 + ((size_t)tid * IN_CH) / 2;      // contiguous 128 u32
    #pragma unroll 4
    for (int k = 0; k < IN_CH; k += 2) {
        unsigned lo = f2bf_bits(src[(size_t)k * HW]);
        unsigned hi = f2bf_bits(src[(size_t)(k + 1) * HW]);
        dst[k >> 1] = lo | (hi << 16);
    }
}

// ---------------------------------------------------------------------------
// Kernel 1: 1x1 conv as bf16 WMMA GEMM, one wave per 16(M) x 64(N) tile.
//   out[b][oc][hw] = sum_c W[oc][c] * X[b][c][hw] + bias[oc]
// A: wbf row-major (256x256), per-lane two 16B loads per K-step.
// B: xbf (B,HW,C) K-contiguous, one 32B load per 16-wide sub-tile.
// 4 accumulators -> 4 WMMAs per K-step, 32 per block; zero VALU converts.
// Epilogue: bias + sigmoid on ch {0,1,4..84}, store to (B,A,y,x,85).
// ---------------------------------------------------------------------------
__global__ __launch_bounds__(32)
void yolo_gemm_wmma(const __bf16* __restrict__ wbf,
                    const __bf16* __restrict__ xbf,
                    const float* __restrict__ cb,
                    float* __restrict__ outp /* CELLS*85, (b,a,hw,ch) */) {
    const int lane = threadIdx.x;          // 0..31
    const int half = lane >> 4;            // 0 or 1
    const int l16  = lane & 15;
    const int nt0  = blockIdx.x * 4;       // first 16-wide n-tile (0..360)
    const int m0   = blockIdx.y * 16;      // oc tile base (0..240)
    const int b    = blockIdx.z;

    // A operand rows (same 16 rows in both halves; K range split by half)
    const __bf16* __restrict__ wrow = wbf + (size_t)(m0 + l16) * IN_CH;

    // B operand columns, one pointer per 16-wide sub-tile (clamped if OOB)
    bool valid[4];
    const __bf16* bptr[4];
    #pragma unroll
    for (int j = 0; j < 4; ++j) {
        int nt = nt0 + j;
        valid[j] = (nt < NTILES);
        int ntc = valid[j] ? nt : (NTILES - 1);
        bptr[j] = xbf + ((size_t)b * HW + ntc * 16 + l16) * IN_CH + 16 * half;
    }

    v8f acc0 = {0.f,0.f,0.f,0.f,0.f,0.f,0.f,0.f};
    v8f acc1 = acc0, acc2 = acc0, acc3 = acc0;

    union AFrag { v16bf v; v8bf h[2]; };

    #pragma unroll
    for (int k0 = 0; k0 < IN_CH; k0 += 32) {
        // A 16x32 bf16 layout: lanes 0-15 -> K 0..15; 16-31 -> K 16..31
        AFrag a;
        a.h[0] = *(const v8bf*)(wrow + k0 + 16 * half);
        a.h[1] = *(const v8bf*)(wrow + k0 + 16 * half + 8);
        // B 32x16 bf16 layout: lanes 0-15 hold K=0..15, lanes 16-31 K=16..31
        v16bf b0 = *(const v16bf*)(bptr[0] + k0);
        v16bf b1 = *(const v16bf*)(bptr[1] + k0);
        v16bf b2 = *(const v16bf*)(bptr[2] + k0);
        v16bf b3 = *(const v16bf*)(bptr[3] + k0);

        acc0 = __builtin_amdgcn_wmma_f32_16x16x32_bf16(false, a.v, false, b0, (short)0, acc0, false, false);
        acc1 = __builtin_amdgcn_wmma_f32_16x16x32_bf16(false, a.v, false, b1, (short)0, acc1, false, false);
        acc2 = __builtin_amdgcn_wmma_f32_16x16x32_bf16(false, a.v, false, b2, (short)0, acc2, false, false);
        acc3 = __builtin_amdgcn_wmma_f32_16x16x32_bf16(false, a.v, false, b3, (short)0, acc3, false, false);
    }

    // C/D layout: lane half selects M rows 0..7 vs 8..15; VGPR r = row offset
    v8f* accs[4] = {&acc0, &acc1, &acc2, &acc3};
    #pragma unroll
    for (int j = 0; j < 4; ++j) {
        if (!valid[j]) continue;
        int hw = (nt0 + j) * 16 + l16;
        #pragma unroll
        for (int r = 0; r < 8; ++r) {
            int oc = m0 + r + 8 * half;
            if (oc < OC) {
                float v = (*accs[j])[r] + cb[oc];
                int a_  = oc / N_CH;
                int ch  = oc - a_ * N_CH;
                if (ch < 2 || ch >= 4) v = sigmoidf(v);
                outp[((size_t)((b * N_ANCH + a_) * HW + hw)) * N_CH + ch] = v;
            }
        }
    }
}

// ---------------------------------------------------------------------------
// Kernel 2: per-label preparation (B*K = 800 items)
// lws layout (16 floats per label):
// [valid, tx, ty, tw, th, ii, jj, best_n, sel, scale, dx, dy, lw, lh, cls, 0]
// ---------------------------------------------------------------------------
__global__ void label_prep(const float* __restrict__ labels, float* __restrict__ lws) {
    int tid = blockIdx.x * blockDim.x + threadIdx.x;
    if (tid >= BATCH * KLAB) return;
    const float* l = labels + (size_t)tid * 5;
    float cls = l[0];
    float sum = l[0] + l[1] + l[2] + l[3] + l[4];
    float valid = (sum > 0.0f) ? 1.0f : 0.0f;
    float tx = l[1] * FSIZE, ty = l[2] * FSIZE;
    float tw = l[3] * FSIZE, th = l[4] * FSIZE;
    int ii = (int)tx, jj = (int)ty;

    // anchor-match IoU (boxes anchored at origin), argmax = first max
    int best = 0; float bestv = -1.0f;
    float tarea = tw * th;
    #pragma unroll
    for (int j = 0; j < 9; ++j) {
        float inter = fminf(tw, AGW[j]) * fminf(th, AGH[j]);
        float iou = inter / (tarea + AGW[j] * AGH[j] - inter);
        if (iou > bestv) { bestv = iou; best = j; }
    }
    int best_n = best % 3;
    float bmask = (best < 3) ? 1.0f : 0.0f;
    float sel = valid * bmask;
    float scale = sqrtf(2.0f - tarea / (float)(HW));
    float dx = tx - (float)ii, dy = ty - (float)jj;
    float lw = logf(tw / MW[best_n] + 1e-16f);
    float lh = logf(th / MH[best_n] + 1e-16f);

    float* o = lws + (size_t)tid * 16;
    o[0] = valid; o[1] = tx; o[2] = ty; o[3] = tw; o[4] = th;
    o[5] = (float)ii; o[6] = (float)jj; o[7] = (float)best_n;
    o[8] = sel; o[9] = scale; o[10] = dx; o[11] = dy;
    o[12] = lw; o[13] = lh; o[14] = cls; o[15] = 0.0f;
}

// ---------------------------------------------------------------------------
// Kernel 3: ignore mask — per cell, max IoU(pred, truths) > 0.7
// ---------------------------------------------------------------------------
__global__ void ignore_mask(const float* __restrict__ outp,
                            const float* __restrict__ lws,
                            float* __restrict__ ignoreF) {
    int idx = blockIdx.x * blockDim.x + threadIdx.x;
    if (idx >= CELLS) return;
    int hw = idx % HW;
    int a  = (idx / HW) % N_ANCH;
    int b  = idx / (HW * N_ANCH);

    const float* o = outp + (size_t)idx * N_CH;
    float gx = (float)(hw % FSIZE), gy = (float)(hw / FSIZE);
    float px = o[0] + gx;
    float py = o[1] + gy;
    float pw = expf(o[2]) * MW[a];
    float ph = expf(o[3]) * MH[a];
    float parea = pw * ph;

    float maxv = 0.0f;
    const float* lb = lws + (size_t)b * KLAB * 16;
    for (int k = 0; k < KLAB; ++k) {
        const float* l = lb + (size_t)k * 16;
        float valid = l[0];
        float tx = l[1], ty = l[2], tw = l[3], th = l[4];
        float tlx = fmaxf(px - pw * 0.5f, tx - tw * 0.5f);
        float tly = fmaxf(py - ph * 0.5f, ty - th * 0.5f);
        float brx = fminf(px + pw * 0.5f, tx + tw * 0.5f);
        float bry = fminf(py + ph * 0.5f, ty + th * 0.5f);
        float en = ((tlx < brx) && (tly < bry)) ? 1.0f : 0.0f;
        float inter = (brx - tlx) * (bry - tly) * en;
        float iou = inter / (parea + tw * th - inter);
        iou *= valid;
        maxv = fmaxf(maxv, iou);
    }
    ignoreF[idx] = (maxv > IGNORE_THRE) ? 1.0f : 0.0f;
}

// ---------------------------------------------------------------------------
// Kernel 4: scatter positive targets into per-cell buffers
// ---------------------------------------------------------------------------
__global__ void scatter_targets(const float* __restrict__ lws,
                                float* __restrict__ posF,
                                float* __restrict__ scaleB,
                                float* __restrict__ tClsB,
                                float* __restrict__ tBox) {
    int tid = blockIdx.x * blockDim.x + threadIdx.x;
    if (tid >= BATCH * KLAB) return;
    int b = tid / KLAB;
    const float* l = lws + (size_t)tid * 16;
    if (l[8] <= 0.5f) return;  // !sel
    int ii = (int)l[5], jj = (int)l[6], bn = (int)l[7];
    if (ii < 0 || ii >= FSIZE || jj < 0 || jj >= FSIZE) return;  // mode='drop'
    int idx = (b * N_ANCH + bn) * HW + jj * FSIZE + ii;
    posF[idx]   = 1.0f;
    scaleB[idx] = l[9];
    tClsB[idx]  = l[14];
    tBox[(size_t)idx * 4 + 0] = l[10];
    tBox[(size_t)idx * 4 + 1] = l[11];
    tBox[(size_t)idx * 4 + 2] = l[12];
    tBox[(size_t)idx * 4 + 3] = l[13];
}

// ---------------------------------------------------------------------------
// Kernel 5: losses + masked out_m (overwrites out in place in d_out)
// d_out scalars: [loss, loss_xy, loss_wh, loss_obj, loss_cls, loss_l2]
// ---------------------------------------------------------------------------
__global__ void loss_kernel(float* __restrict__ outp,
                            const float* __restrict__ posF,
                            const float* __restrict__ scaleB,
                            const float* __restrict__ tClsB,
                            const float* __restrict__ tBox,
                            const float* __restrict__ ignoreF,
                            float* __restrict__ scalars) {
    int idx = blockIdx.x * blockDim.x + threadIdx.x;
    float lxy = 0.f, lwh = 0.f, lobj = 0.f, lcls = 0.f, l2 = 0.f;

    if (idx < CELLS) {
        float* o   = outp + (size_t)idx * N_CH;
        float p    = posF[idx];
        float s    = scaleB[idx];
        float g    = ignoreF[idx];
        float obj  = (p > 0.5f) ? 1.0f : (1.0f - g);
        int   cid  = (int)tClsB[idx];
        float w2   = s * s;

        // xy (ch 0,1): bce weighted by tgt_scale^2
        #pragma unroll
        for (int c = 0; c < 2; ++c) {
            float ov = o[c] * p;
            float tv = tBox[(size_t)idx * 4 + c] * p;
            lxy += bce1(ov, tv) * w2;
            float d = ov - tv; l2 += d * d;
            o[c] = ov;
        }
        // wh (ch 2,3): 0.5*sum((o-t)^2), both scaled by tgt_scale
        #pragma unroll
        for (int c = 2; c < 4; ++c) {
            float ov = o[c] * p * s;
            float tv = tBox[(size_t)idx * 4 + c] * p * s;
            float d = ov - tv;
            lwh += 0.5f * d * d;
            l2  += d * d;
            o[c] = ov;
        }
        // obj (ch 4)
        {
            float ov = o[4] * obj;
            float tv = p;   // target obj = 1 at positives; obj_mask=1 there
            lobj += bce1(ov, tv);
            float d = ov - tv; l2 += d * d;
            o[4] = ov;
        }
        // cls (ch 5..84)
        for (int c = 0; c < N_CLASSES; ++c) {
            float ov = o[5 + c] * p;
            float tv = (p > 0.5f && c == cid) ? 1.0f : 0.0f;
            lcls += bce1(ov, tv);
            float d = ov - tv; l2 += d * d;
            o[5 + c] = ov;
        }
    }

    lxy  = wave_red(lxy);
    lwh  = wave_red(lwh);
    lobj = wave_red(lobj);
    lcls = wave_red(lcls);
    l2   = wave_red(l2);
    if ((threadIdx.x & 31) == 0) {
        atomicAdd(&scalars[1], lxy);
        atomicAdd(&scalars[2], lwh);
        atomicAdd(&scalars[3], lobj);
        atomicAdd(&scalars[4], lcls);
        atomicAdd(&scalars[5], l2);
        atomicAdd(&scalars[0], lxy + lwh + lobj + lcls);
    }
}

// ---------------------------------------------------------------------------
// Launch
// ---------------------------------------------------------------------------
extern "C" void kernel_launch(void* const* d_in, const int* in_sizes, int n_in,
                              void* d_out, int out_size, void* d_ws, size_t ws_size,
                              hipStream_t stream) {
    const float* xin    = (const float*)d_in[0];   // (16,256,76,76)
    const float* labels = (const float*)d_in[1];   // (16,50,5)
    const float* cw     = (const float*)d_in[2];   // (255,256)
    const float* cb     = (const float*)d_in[3];   // (255,)

    float* out  = (float*)d_out;       // 6 scalars + out_m (B,A,76,76,85)
    float* outm = out + 6;

    // workspace layout (floats unless noted):
    //   [0,C)      ignoreF          (fully overwritten each call)
    //   [C,2C)     posF             (memset 0)
    //   [2C,3C)    scaleB           (memset 0)
    //   [3C,4C)    tClsB            (memset 0)
    //   [4C,8C)    tBox             (memset 0)
    //   [8C,..)    lws  800*16      (fully overwritten)
    //   then       wbf  256*256 bf16 (128 KB)
    //   then       xbf  B*HW*256 bf16 (47.3 MB)
    float* ws      = (float*)d_ws;
    float* ignoreF = ws;
    float* posF    = ws + (size_t)CELLS;
    float* scaleB  = ws + (size_t)2 * CELLS;
    float* tClsB   = ws + (size_t)3 * CELLS;
    float* tBox    = ws + (size_t)4 * CELLS;
    float* lws     = ws + (size_t)8 * CELLS;
    size_t lwsEnd  = (size_t)8 * CELLS + (size_t)BATCH * KLAB * 16;   // in floats
    unsigned short* wbf = (unsigned short*)(ws + lwsEnd);
    unsigned short* xbf = wbf + (size_t)256 * IN_CH;

    // zero scatter buffers + loss scalars
    hipMemsetAsync(posF, 0, (size_t)7 * CELLS * sizeof(float), stream);
    hipMemsetAsync(d_out, 0, 6 * sizeof(float), stream);

    // 0) one-time bf16 conversions (W padded to 256 rows; X transposed to (B,HW,C))
    conv_w_to_bf16<<<(256 * IN_CH + 255) / 256, 256, 0, stream>>>(cw, wbf);
    xin_to_bf16_t<<<(BATCH * HW + 255) / 256, 256, 0, stream>>>(xin, (unsigned*)xbf);

    // 1) GEMM: one wave per 16x64 tile; grid = (ceil(361/4), 16 m-tiles, 16 batch)
    dim3 ggrid((NTILES + 3) / 4, 16, BATCH);   // 91 x 16 x 16
    yolo_gemm_wmma<<<ggrid, 32, 0, stream>>>((const __bf16*)wbf, (const __bf16*)xbf, cb, outm);

    // 2) labels
    label_prep<<<(BATCH * KLAB + 255) / 256, 256, 0, stream>>>(labels, lws);

    // 3) ignore mask
    ignore_mask<<<(CELLS + 255) / 256, 256, 0, stream>>>(outm, lws, ignoreF);

    // 4) scatter positives
    scatter_targets<<<(BATCH * KLAB + 255) / 256, 256, 0, stream>>>(
        lws, posF, scaleB, tClsB, tBox);

    // 5) losses + masked out_m (in place)
    loss_kernel<<<(CELLS + 255) / 256, 256, 0, stream>>>(
        outm, posF, scaleB, tClsB, tBox, ignoreF, out);
}
